// NONLocalBlock2D_30511447671193
// MI455X (gfx1250) — matrix operand order repaired and checked by reading, not compiled, not run on hardware
//
#include <hip/hip_runtime.h>
#include <cstddef>

// ---------------------------------------------------------------------------
// NonLocal block (attention) for MI455X / gfx1250.
// All GEMMs (1x1 convs, q^T k, P v, output conv) run on v_wmma_f32_16x16x32_bf16.
// Attention is fused flash-style (online softmax) so the 268MB sim matrix never
// touches HBM. fp32 accumulation everywhere; bf16 operands.
// Round 2: software-pipelined fragment loads (register ping-pong) in the GEMMs,
// 32x64 per-wave tiles, and load/compute overlap + K-prefetch in flash.
// ---------------------------------------------------------------------------

typedef __attribute__((ext_vector_type(16))) __bf16 v16bf;
typedef __attribute__((ext_vector_type(8)))  float  v8f;

union Frag {
    v16bf v;
    uint4 u[2];
};

static __device__ __forceinline__ v8f wmma_bf16(const Frag& a, const Frag& b, v8f c) {
    return __builtin_amdgcn_wmma_f32_16x16x32_bf16(
        /*neg_a=*/false, a.v, /*neg_b=*/false, b.v,
        /*c_mod=*/(short)0, c, /*reuse_a=*/false, /*reuse_b=*/false);
}

// float -> bf16 (round to nearest even), stored as ushort
static __device__ __forceinline__ unsigned short f2bf(float f) {
    unsigned int u = __float_as_uint(f);
    u += 0x7FFFu + ((u >> 16) & 1u);
    return (unsigned short)(u >> 16);
}

// problem constants
#define NB   4
#define CIN  256
#define CK   128
#define CVD  128
#define COUT 256
#define PHW  4096

// ---------------------------------------------------------------------------
// prep kernels
// ---------------------------------------------------------------------------
__global__ void bn_prep_kernel(const float* __restrict__ g, const float* __restrict__ b,
                               const float* __restrict__ m, const float* __restrict__ v,
                               float* __restrict__ scale, float* __restrict__ shift, int n) {
    int i = blockIdx.x * blockDim.x + threadIdx.x;
    if (i < n) {
        float sc = g[i] * rsqrtf(v[i] + 1e-5f);
        scale[i] = sc;
        shift[i] = b[i] - m[i] * sc;
    }
}

__global__ void cvt_bf16_kernel(const float* __restrict__ src, unsigned short* __restrict__ dst, int n) {
    int i = blockIdx.x * blockDim.x + threadIdx.x;
    if (i < n) dst[i] = f2bf(src[i]);
}

// [N][C][P] f32  ->  [N][P][C] bf16   (LDS tiled transpose, block 32x8)
__global__ void transpose_cvt_kernel(const float* __restrict__ x, unsigned short* __restrict__ xt, int Cdim) {
    __shared__ unsigned short tile[32][33];
    int p0 = blockIdx.x * 32, c0 = blockIdx.y * 32, n = blockIdx.z;
    int tx = threadIdx.x, ty = threadIdx.y;
#pragma unroll
    for (int i = 0; i < 4; i++) {
        int c = c0 + ty + i * 8;
        tile[ty + i * 8][tx] = f2bf(x[((size_t)n * Cdim + c) * PHW + p0 + tx]);
    }
    __syncthreads();
#pragma unroll
    for (int i = 0; i < 4; i++) {
        int p = p0 + ty + i * 8;
        xt[((size_t)n * PHW + p) * Cdim + c0 + tx] = tile[tx][ty + i * 8];
    }
}

// ---------------------------------------------------------------------------
// WMMA GEMM: Out = act(scale * (W @ X) + shift)
//  W  : [Mdim][Kdim] bf16 row-major
//  Xt : [N][PHW][Kdim] bf16 (spatial-major activations)
//  Out: trans_out ? [N][PHW][Mdim] : [N][Mdim][PHW]   (bf16)
//  scale==nullptr -> plain GEMM (no BN, no ReLU)
// grid: (PHW/128, Mdim/128, N); block 256.
// 8 waves in 4(M)x2(N): each wave computes a 32x64 tile, 8 WMMAs per k-step,
// with register ping-pong double buffering of fragments (Kdim % 64 == 0).
// ---------------------------------------------------------------------------
struct GSet {
    Frag a[2];
    Frag b[4];
};

__global__ __launch_bounds__(256) void gemm_bn_relu_kernel(
    const unsigned short* __restrict__ Wb, const unsigned short* __restrict__ Xt,
    unsigned short* __restrict__ Out, const float* __restrict__ scale,
    const float* __restrict__ shift, int Mdim, int Kdim, int trans_out)
{
    int lane = threadIdx.x & 31, wave = threadIdx.x >> 5;
    int half = lane >> 4, lm = lane & 15;
    int wm = wave & 3, wn = wave >> 2;
    int n  = blockIdx.z;
    int m0 = blockIdx.y * 128 + wm * 32;
    int p0 = blockIdx.x * 128 + wn * 64;

    v8f acc[2][4] = {};

    const unsigned short* arow0 = Wb + (size_t)(m0 + lm) * Kdim + half * 8;
    const unsigned short* arow1 = arow0 + (size_t)16 * Kdim;
    const unsigned short* brow[4];
#pragma unroll
    for (int t = 0; t < 4; t++)
        brow[t] = Xt + ((size_t)n * PHW + (p0 + t * 16 + lm)) * Kdim + half * 8;

    auto loadSet = [&](GSet& s, int k0) {
        s.a[0].u[0] = *reinterpret_cast<const uint4*>(arow0 + k0);
        s.a[0].u[1] = *reinterpret_cast<const uint4*>(arow0 + k0 + 16);
        s.a[1].u[0] = *reinterpret_cast<const uint4*>(arow1 + k0);
        s.a[1].u[1] = *reinterpret_cast<const uint4*>(arow1 + k0 + 16);
#pragma unroll
        for (int t = 0; t < 4; t++) {
            s.b[t].u[0] = *reinterpret_cast<const uint4*>(brow[t] + k0);
            s.b[t].u[1] = *reinterpret_cast<const uint4*>(brow[t] + k0 + 16);
        }
    };
    auto mmaSet = [&](GSet& s) {
#pragma unroll
        for (int ai = 0; ai < 2; ai++)
#pragma unroll
            for (int t = 0; t < 4; t++)
                acc[ai][t] = wmma_bf16(s.a[ai], s.b[t], acc[ai][t]);
    };

    GSet s0, s1;
    loadSet(s0, 0);
    for (int k0 = 0; k0 < Kdim; k0 += 64) {
        loadSet(s1, k0 + 32);   // in flight while s0 WMMAs run
        mmaSet(s0);
        if (k0 + 64 < Kdim) loadSet(s0, k0 + 64);
        mmaSet(s1);
    }

    bool act = (scale != nullptr);
#pragma unroll
    for (int ai = 0; ai < 2; ai++) {
#pragma unroll
        for (int r = 0; r < 8; r++) {
            int m = m0 + ai * 16 + r + half * 8;
            float sc = 1.f, sh = 0.f;
            if (act) { sc = scale[m]; sh = shift[m]; }
#pragma unroll
            for (int t = 0; t < 4; t++) {
                int p = p0 + t * 16 + lm;
                float val = acc[ai][t][r];
                if (act) val = fmaxf(val * sc + sh, 0.f);
                unsigned short bv = f2bf(val);
                if (trans_out) Out[((size_t)n * PHW + p) * Mdim + m] = bv;
                else           Out[((size_t)n * Mdim + m) * PHW + p] = bv;
            }
        }
    }
}

// ---------------------------------------------------------------------------
// Flash attention: ctx^T[p][cv] = softmax_k( q^T k / sqrt(Ck) ) @ v^T
//  Q  : [N][PHW][CK]  bf16 (queries, spatial-major)
//  Kt : [N][PHW][CK]  bf16 (keys, spatial-major)
//  V  : [N][CVD][PHW] bf16 (values, channel-major)
//  CtxT: [N][PHW][CVD] bf16
// One wave owns 16 queries; loops all 4096 keys in chunks of 32.
// K fragments are issued as a batch ahead of the S WMMAs; V fragments are
// issued before the softmax so exp/shfl VALU work hides their latency.
// grid: (N*PHW/16/8); block 256 (8 waves)
// ---------------------------------------------------------------------------
__global__ __launch_bounds__(256) void flash_attn_kernel(
    const unsigned short* __restrict__ Q, const unsigned short* __restrict__ Kt,
    const unsigned short* __restrict__ V, unsigned short* __restrict__ CtxT)
{
    __shared__ unsigned short ptile[8 * 512];   // per-wave 16x32 bf16 P tile
    int lane = threadIdx.x & 31, wave = threadIdx.x >> 5;
    int half = lane >> 4, lm = lane & 15;
    int strip = blockIdx.x * 8 + wave;          // 1024 strips of 16 queries
    int n  = strip >> 8;
    int q0 = (strip & 255) << 4;
    const float sm = 0.08838834764831845f;      // 1/sqrt(128)

    // Q A-fragments for the 4 k-steps (c = 0..127), resident in registers
    Frag qa[4];
#pragma unroll
    for (int j = 0; j < 4; j++) {
        const unsigned short* qrow = Q + ((size_t)n * PHW + q0 + lm) * CK + j * 32 + half * 16;
        qa[j].u[0] = *reinterpret_cast<const uint4*>(qrow);
        qa[j].u[1] = *reinterpret_cast<const uint4*>(qrow + 8);
    }

    v8f o[8] = {};
    float mrow[8], srow[8];
#pragma unroll
    for (int r = 0; r < 8; r++) { mrow[r] = -1e30f; srow[r] = 0.f; }

    unsigned short* pl = ptile + wave * 512;
    const unsigned short* ktbase = Kt + (size_t)n * PHW * CK;
    const unsigned short* vbase  = V + (size_t)n * CVD * PHW;

    for (int k0 = 0; k0 < PHW; k0 += 32) {
        // prefetch next chunk's K rows (fresh 256B rows; V bytes for the next
        // chunk live on lines already pulled by this chunk's V loads)
        if (k0 + 32 < PHW) {
            __builtin_prefetch(ktbase + (size_t)(k0 + 32 + lm) * CK, 0, 1);
            __builtin_prefetch(ktbase + (size_t)(k0 + 48 + lm) * CK, 0, 1);
        }

        // batch-issue all 8 K fragments for this chunk
        Frag kf[8];
#pragma unroll
        for (int j = 0; j < 4; j++) {
            const unsigned short* kr0 = ktbase + (size_t)(k0 + lm) * CK + j * 32 + half * 16;
            const unsigned short* kr1 = kr0 + (size_t)16 * CK;
            kf[2 * j].u[0]     = *reinterpret_cast<const uint4*>(kr0);
            kf[2 * j].u[1]     = *reinterpret_cast<const uint4*>(kr0 + 8);
            kf[2 * j + 1].u[0] = *reinterpret_cast<const uint4*>(kr1);
            kf[2 * j + 1].u[1] = *reinterpret_cast<const uint4*>(kr1 + 8);
        }

        // S = q^T k for two 16-key tiles
        v8f s0v = {}, s1v = {};
#pragma unroll
        for (int j = 0; j < 4; j++) {
            s0v = wmma_bf16(qa[j], kf[2 * j], s0v);
            s1v = wmma_bf16(qa[j], kf[2 * j + 1], s1v);
        }

        // issue V fragments now; the softmax below hides their latency
        Frag vf[8];
#pragma unroll
        for (int t = 0; t < 8; t++) {
            const unsigned short* vr = vbase + (size_t)(t * 16 + lm) * PHW + k0 + half * 16;
            vf[t].u[0] = *reinterpret_cast<const uint4*>(vr);
            vf[t].u[1] = *reinterpret_cast<const uint4*>(vr + 8);
        }

        // online softmax per query row (rows live across the 16-lane half)
#pragma unroll
        for (int r = 0; r < 8; r++) {
            float x0 = s0v[r] * sm, x1 = s1v[r] * sm;
            float mx = fmaxf(x0, x1);
            mx = fmaxf(mx, __shfl_xor(mx, 1));
            mx = fmaxf(mx, __shfl_xor(mx, 2));
            mx = fmaxf(mx, __shfl_xor(mx, 4));
            mx = fmaxf(mx, __shfl_xor(mx, 8));
            float mnew = fmaxf(mrow[r], mx);
            float corr = __expf(mrow[r] - mnew);
            float p0 = __expf(x0 - mnew);
            float p1 = __expf(x1 - mnew);
            float rs = p0 + p1;
            rs += __shfl_xor(rs, 1);
            rs += __shfl_xor(rs, 2);
            rs += __shfl_xor(rs, 4);
            rs += __shfl_xor(rs, 8);
            srow[r] = srow[r] * corr + rs;
            mrow[r] = mnew;
#pragma unroll
            for (int t = 0; t < 8; t++) o[t][r] *= corr;
            // D-layout -> LDS (query, key) tile
            pl[(r + 8 * half) * 32 + lm]      = f2bf(p0);
            pl[(r + 8 * half) * 32 + 16 + lm] = f2bf(p1);
        }

        // reload P in A-fragment layout (wave-private LDS region, in-order ds)
        Frag pa;
        pa.u[0] = *reinterpret_cast<const uint4*>(pl + lm * 32 + half * 16);
        pa.u[1] = *reinterpret_cast<const uint4*>(pl + lm * 32 + half * 16 + 8);

        // O += P @ V over 8 cv tiles (V already in registers)
#pragma unroll
        for (int t = 0; t < 8; t++)
            o[t] = wmma_bf16(pa, vf[t], o[t]);
    }

    // finalize: divide by row sum, store ctx^T [N][PHW][CVD]
#pragma unroll
    for (int r = 0; r < 8; r++) {
        float inv = 1.f / srow[r];
        int p = q0 + r + 8 * half;
        unsigned short* orow = CtxT + ((size_t)n * PHW + p) * CVD;
#pragma unroll
        for (int t = 0; t < 8; t++) orow[t * 16 + lm] = f2bf(o[t][r] * inv);
    }
}

// ---------------------------------------------------------------------------
// Output GEMM: out = x + relu(scale*(Wo @ ctx) + shift), fp32 output
// grid: (PHW/128, COUT/128, N); block 256; same pipelined 32x64 wave tiles
// ---------------------------------------------------------------------------
__global__ __launch_bounds__(256) void gemm_out_residual_kernel(
    const unsigned short* __restrict__ Wb, const unsigned short* __restrict__ Xt,
    const float* __restrict__ scale, const float* __restrict__ shift,
    const float* __restrict__ resid, float* __restrict__ Out)
{
    const int Kdim = CVD;
    int lane = threadIdx.x & 31, wave = threadIdx.x >> 5;
    int half = lane >> 4, lm = lane & 15;
    int wm = wave & 3, wn = wave >> 2;
    int n  = blockIdx.z;
    int m0 = blockIdx.y * 128 + wm * 32;
    int p0 = blockIdx.x * 128 + wn * 64;

    v8f acc[2][4] = {};

    const unsigned short* arow0 = Wb + (size_t)(m0 + lm) * Kdim + half * 8;
    const unsigned short* arow1 = arow0 + (size_t)16 * Kdim;
    const unsigned short* brow[4];
#pragma unroll
    for (int t = 0; t < 4; t++)
        brow[t] = Xt + ((size_t)n * PHW + (p0 + t * 16 + lm)) * Kdim + half * 8;

    auto loadSet = [&](GSet& s, int k0) {
        s.a[0].u[0] = *reinterpret_cast<const uint4*>(arow0 + k0);
        s.a[0].u[1] = *reinterpret_cast<const uint4*>(arow0 + k0 + 16);
        s.a[1].u[0] = *reinterpret_cast<const uint4*>(arow1 + k0);
        s.a[1].u[1] = *reinterpret_cast<const uint4*>(arow1 + k0 + 16);
#pragma unroll
        for (int t = 0; t < 4; t++) {
            s.b[t].u[0] = *reinterpret_cast<const uint4*>(brow[t] + k0);
            s.b[t].u[1] = *reinterpret_cast<const uint4*>(brow[t] + k0 + 16);
        }
    };
    auto mmaSet = [&](GSet& s) {
#pragma unroll
        for (int ai = 0; ai < 2; ai++)
#pragma unroll
            for (int t = 0; t < 4; t++)
                acc[ai][t] = wmma_bf16(s.a[ai], s.b[t], acc[ai][t]);
    };

    GSet s0, s1;
    loadSet(s0, 0);
    for (int k0 = 0; k0 < Kdim; k0 += 64) {
        loadSet(s1, k0 + 32);
        mmaSet(s0);
        if (k0 + 64 < Kdim) loadSet(s0, k0 + 64);
        mmaSet(s1);
    }

#pragma unroll
    for (int ai = 0; ai < 2; ai++) {
#pragma unroll
        for (int r = 0; r < 8; r++) {
            int m = m0 + ai * 16 + r + half * 8;
            float sc = scale[m], sh = shift[m];
#pragma unroll
            for (int t = 0; t < 4; t++) {
                int p = p0 + t * 16 + lm;
                size_t idx = ((size_t)n * COUT + m) * PHW + p;
                Out[idx] = resid[idx] + fmaxf(acc[ai][t][r] * sc + sh, 0.f);
            }
        }
    }
}

// ---------------------------------------------------------------------------
// launch
// ---------------------------------------------------------------------------
extern "C" void kernel_launch(void* const* d_in, const int* in_sizes, int n_in,
                              void* d_out, int out_size, void* d_ws, size_t ws_size,
                              hipStream_t stream) {
    (void)in_sizes; (void)n_in; (void)out_size;
    const float* x   = (const float*)d_in[0];
    const float* y   = (const float*)d_in[1];
    const float* Wq1 = (const float*)d_in[2];
    const float* Wq2 = (const float*)d_in[3];
    const float* Wk1 = (const float*)d_in[4];
    const float* Wk2 = (const float*)d_in[5];
    const float* Wv  = (const float*)d_in[6];
    const float* Wo  = (const float*)d_in[7];
    const float* bn[5][4]; // q1,q2,k1,k2,o : g,b,m,v
    for (int s = 0; s < 5; s++)
        for (int j = 0; j < 4; j++)
            bn[s][j] = (const float*)d_in[8 + s * 4 + j];
    float* out = (float*)d_out;

    char* ws = (char*)d_ws;
    size_t off = 0;
    auto take = [&](size_t bytes) -> void* {
        void* p = ws + off;
        off += (bytes + 255) & ~(size_t)255;
        return p;
    };
    unsigned short* wq1b = (unsigned short*)take(CK * CIN * 2);
    unsigned short* wq2b = (unsigned short*)take(CK * CK * 2);
    unsigned short* wk1b = (unsigned short*)take(CK * CIN * 2);
    unsigned short* wk2b = (unsigned short*)take(CK * CK * 2);
    unsigned short* wvb  = (unsigned short*)take(CVD * CIN * 2);
    unsigned short* wob  = (unsigned short*)take(COUT * CVD * 2);
    float* bnbuf[5][2]; // scale, shift
    int bnc[5] = {CK, CK, CK, CK, COUT};
    for (int s = 0; s < 5; s++) {
        bnbuf[s][0] = (float*)take(bnc[s] * 4);
        bnbuf[s][1] = (float*)take(bnc[s] * 4);
    }
    unsigned short* XT   = (unsigned short*)take((size_t)NB * PHW * CIN * 2);
    unsigned short* YT   = (unsigned short*)take((size_t)NB * PHW * CIN * 2);
    unsigned short* Q1T  = (unsigned short*)take((size_t)NB * PHW * CK * 2);
    unsigned short* K1T  = (unsigned short*)take((size_t)NB * PHW * CK * 2);
    unsigned short* QT   = (unsigned short*)take((size_t)NB * PHW * CK * 2);
    unsigned short* KT   = (unsigned short*)take((size_t)NB * PHW * CK * 2);
    unsigned short* Vn   = (unsigned short*)take((size_t)NB * CVD * PHW * 2);
    unsigned short* CTXT = (unsigned short*)take((size_t)NB * PHW * CVD * 2);
    if (off > ws_size) return; // workspace too small: bail deterministically

    // --- BN fold + weight conversion ---
    for (int s = 0; s < 5; s++)
        bn_prep_kernel<<<dim3((bnc[s] + 255) / 256), 256, 0, stream>>>(
            bn[s][0], bn[s][1], bn[s][2], bn[s][3], bnbuf[s][0], bnbuf[s][1], bnc[s]);
    struct { const float* s; unsigned short* d; int n; } wc[6] = {
        {Wq1, wq1b, CK * CIN}, {Wq2, wq2b, CK * CK}, {Wk1, wk1b, CK * CIN},
        {Wk2, wk2b, CK * CK},  {Wv, wvb, CVD * CIN}, {Wo, wob, COUT * CVD}};
    for (int i = 0; i < 6; i++)
        cvt_bf16_kernel<<<dim3((wc[i].n + 255) / 256), 256, 0, stream>>>(wc[i].s, wc[i].d, wc[i].n);

    // --- activations -> spatial-major bf16 ---
    transpose_cvt_kernel<<<dim3(PHW / 32, CIN / 32, NB), dim3(32, 8), 0, stream>>>(x, XT, CIN);
    transpose_cvt_kernel<<<dim3(PHW / 32, CIN / 32, NB), dim3(32, 8), 0, stream>>>(y, YT, CIN);

    // --- projection GEMMs (conv1x1 + BN + ReLU fused) ---
    dim3 gg(PHW / 128, 1, NB);
    gemm_bn_relu_kernel<<<gg, 256, 0, stream>>>(wq1b, XT,  Q1T, bnbuf[0][0], bnbuf[0][1], CK, CIN, 1);
    gemm_bn_relu_kernel<<<gg, 256, 0, stream>>>(wk1b, YT,  K1T, bnbuf[2][0], bnbuf[2][1], CK, CIN, 1);
    gemm_bn_relu_kernel<<<gg, 256, 0, stream>>>(wq2b, Q1T, QT,  bnbuf[1][0], bnbuf[1][1], CK, CK, 1);
    gemm_bn_relu_kernel<<<gg, 256, 0, stream>>>(wk2b, K1T, KT,  bnbuf[3][0], bnbuf[3][1], CK, CK, 1);
    gemm_bn_relu_kernel<<<gg, 256, 0, stream>>>(wvb,  YT,  Vn,  nullptr, nullptr, CVD, CIN, 0);

    // --- fused flash attention ---
    flash_attn_kernel<<<dim3(NB * (PHW / 16) / 8), 256, 0, stream>>>(QT, KT, Vn, CTXT);

    // --- output conv + BN + ReLU + residual ---
    gemm_out_residual_kernel<<<dim3(PHW / 128, COUT / 128, NB), 256, 0, stream>>>(
        wob, CTXT, bnbuf[4][0], bnbuf[4][1], x, out);
}
